// WindowAttention_72000831750669
// MI455X (gfx1250) — matrix operand order repaired and checked by reading, not compiled, not run on hardware
//
#include <hip/hip_runtime.h>

// ---------------------------------------------------------------------------
// Window attention for MI455X (gfx1250), wave32 + v_wmma_f32_16x16x32_bf16.
// One 128-thread block (4 waves, scalarized wave id) per window.
// All fragment traffic is 16-byte (global_load_b128 / ds_load_b128):
//  - weights pre-swizzled to B-fragment order in workspace
//  - V stored transposed in LDS so PV B-fragments are K-contiguous
//  - S computed transposed (A=k, B=q) so the bias/mask/score epilogue is
//    fully vectorized (padded bias in ws, mask staged into LDS stride-64)
// ---------------------------------------------------------------------------

typedef __attribute__((ext_vector_type(16))) __bf16 v16bf;
typedef __attribute__((ext_vector_type(8)))  float  v8f;

union FragBF { v16bf v; uint4 q[2]; unsigned short h[16]; };

#define NTOK 49
#define NPAD 64
#define CDIM 128
#define QKVD 384
#define NN   (NTOK * NTOK)          // 2401
// workspace layout (in ushorts): [0, 96*512)  qkv_w fragments
//                                [49152, 49152+32*512) proj_w fragments
//                                then (float) 4*64*64 padded expanded bias
#define WS_PROJ_OFF  (96 * 512)
#define WS_BIAS_OFF  (WS_PROJ_OFF + 32 * 512)   // ushort offset; *2 = byte off

// Native f32 -> bf16 (round-to-nearest-even) via hardware cvt.
__device__ __forceinline__ unsigned short f2bf(float f) {
  return __builtin_bit_cast(unsigned short, (__bf16)f);
}

// A-matrix 16x32 bf16 fragment: lane holds row m=lane&15; K runs kh..kh+15
// (kh = 16*(lane>>4)) -> two contiguous 16B chunks.
__device__ __forceinline__ v16bf frag_a(const unsigned short* base, int stride, int lane) {
  const int m  = lane & 15;
  const int kh = (lane >> 4) << 4;
  const unsigned short* r = base + m * stride + kh;
  FragBF f;
  f.q[0] = *(const uint4*)(r);
  f.q[1] = *(const uint4*)(r + 8);
  return f.v;
}

// B-matrix 32x16 fragment where logical B(k,n) = mem[n*stride + k]:
// per lane 16 contiguous bf16 at k = 16*(lane>>4).
__device__ __forceinline__ v16bf frag_bt(const unsigned short* base, int stride, int lane) {
  const int n  = lane & 15;
  const int kh = (lane >> 4) << 4;
  const unsigned short* r = base + n * stride + kh;
  FragBF f;
  f.q[0] = *(const uint4*)(r);
  f.q[1] = *(const uint4*)(r + 8);
  return f.v;
}

// B fragment from pre-swizzled global weights: lane's 16 values contiguous.
__device__ __forceinline__ v16bf frag_g(const unsigned short* base, int lane) {
  const unsigned short* r = base + lane * 16;
  FragBF f;
  f.q[0] = *(const uint4*)(r);
  f.q[1] = *(const uint4*)(r + 8);
  return f.v;
}

__device__ __forceinline__ v8f wmma_bf16(v16bf a, v16bf b, v8f c) {
  return __builtin_amdgcn_wmma_f32_16x16x32_bf16(false, a, false, b, (short)0, c, false, false);
}

// ---- pre-pass 1: swizzle weights into B-fragment order (bf16) -------------
__global__ void swizzle_weights_kernel(const float* __restrict__ qkv_w,
                                       const float* __restrict__ proj_w,
                                       unsigned short* __restrict__ ws) {
  const int t    = blockIdx.x * 256 + threadIdx.x;
  const int lane = t & 31;
  const int tile = t >> 5;
  const int n  = lane & 15;
  const int kh = (lane >> 4) << 4;
  if (tile < 96) {
    const int kt = tile / 24, nt = tile % 24;
    unsigned short* dst = ws + (tile * 32 + lane) * 16;
#pragma unroll
    for (int e = 0; e < 16; ++e)
      dst[e] = f2bf(qkv_w[(kt * 32 + kh + e) * QKVD + nt * 16 + n]);
  } else if (tile < 128) {
    const int t2 = tile - 96;
    const int kt = t2 >> 3, nt = t2 & 7;
    unsigned short* dst = ws + WS_PROJ_OFF + (t2 * 32 + lane) * 16;
#pragma unroll
    for (int e = 0; e < 16; ++e)
      dst[e] = f2bf(proj_w[(kt * 32 + kh + e) * CDIM + nt * 16 + n]);
  }
}

// ---- pre-pass 2: expand relative-position bias, padded to [4][64][64] -----
__global__ void bias_expand_kernel(const float* __restrict__ bias_table,
                                   const int* __restrict__ rel_index,
                                   float* __restrict__ biasp) {
  const int i = blockIdx.x * 256 + threadIdx.x;
  if (i < 4 * NPAD * NPAD) {
    const int h = i >> 12, rem = i & 4095;
    const int r = rem >> 6, c = rem & 63;
    biasp[i] = (r < NTOK && c < NTOK)
                   ? bias_table[rel_index[r * NTOK + c] * 4 + h] : 0.0f;
  }
}

__global__ __launch_bounds__(128) void win_attn_kernel(
    const float* __restrict__ x, const float* __restrict__ mask,
    const float* __restrict__ qkv_b, const float* __restrict__ proj_b,
    const unsigned short* __restrict__ wbf, const float* __restrict__ biasp,
    float* __restrict__ out, int nmask) {
  __shared__ alignas(16) unsigned short sh_x[NPAD * CDIM];   // 16 KB (-> mask)
  __shared__ alignas(16) unsigned short sh_qk[NPAD * 256];   // 32 KB q|k
  __shared__ alignas(16) unsigned short sh_vT[CDIM * NPAD];  // 16 KB v^T
  __shared__ alignas(16) float          sh_S[NPAD * NPAD];   // 16 KB scores
  __shared__ alignas(16) unsigned short sh_P[NPAD * NPAD];   //  8 KB probs
  __shared__ alignas(16) unsigned short sh_O[NPAD * CDIM];   // 16 KB attn out
  __shared__ float sh_inv[NPAD];

  const int tid  = threadIdx.x;
  const int wave = __builtin_amdgcn_readfirstlane(tid >> 5);  // scalar wave id
  const int lane = tid & 31;
  const int wi   = blockIdx.x;
  const int mw   = wi % nmask;

  const unsigned short* qkvw_bf  = wbf;
  const unsigned short* projw_bf = wbf + WS_PROJ_OFF;

  // ---- Phase 0: stage x into LDS as bf16 (float4 in, packed uint2 out) ----
  {
    const float4* xv = (const float4*)(x + (size_t)wi * (NTOK * CDIM));
    uint2* dst = (uint2*)sh_x;
    for (int i = tid; i < NPAD * CDIM / 4; i += 128) {
      const int row = i >> 5;  // (i*4)/128
      const float4 v = (row < NTOK) ? xv[i] : make_float4(0.f, 0.f, 0.f, 0.f);
      uint2 pk;
      pk.x = (unsigned)f2bf(v.x) | ((unsigned)f2bf(v.y) << 16);
      pk.y = (unsigned)f2bf(v.z) | ((unsigned)f2bf(v.w) << 16);
      dst[i] = pk;
    }
  }
  __syncthreads();

  // ---- Phase 1: qkv = x @ qkv_w + qkv_b  (64x128 @ 128x384) ----
  for (int nt = wave; nt < 24; nt += 4) {
    v16bf bfr[4];
#pragma unroll
    for (int kk = 0; kk < 4; ++kk)
      bfr[kk] = frag_g(qkvw_bf + (kk * 24 + nt) * 512, lane);
#pragma unroll
    for (int mt = 0; mt < 4; ++mt) {
      v8f acc = {};
#pragma unroll
      for (int kk = 0; kk < 4; ++kk) {
        v16bf a = frag_a(sh_x + (mt * 16) * CDIM + kk * 32, CDIM, lane);
        acc = wmma_bf16(a, bfr[kk], acc);
      }
      const int n = lane & 15, kh = lane >> 4;
      const int gc = nt * 16 + n;
      const float bb = qkv_b[gc];
      if (nt < 16) {   // q|k -> row-major (nt is scalar: s_cbranch)
#pragma unroll
        for (int r = 0; r < 8; ++r) {
          const int row = mt * 16 + r + 8 * kh;
          sh_qk[row * 256 + gc] = f2bf(acc[r] + bb);
        }
      } else {         // v -> transposed [d][token] so PV B-frags are K-contig
        const int vc = gc - 256;
#pragma unroll
        for (int r = 0; r < 8; ++r) {
          const int row = mt * 16 + r + 8 * kh;
          sh_vT[vc * NPAD + row] = f2bf(acc[r] + bb);
        }
      }
    }
  }
  __syncthreads();

  // ---- Phase 1.5: stage window mask into LDS (stride 64, pads = -1e30),
  //      reusing the dead x staging buffer (exactly 4096 floats).
  float* sh_mask = (float*)sh_x;
  {
    const float* maskw = mask + (size_t)mw * NN;
    for (int i = tid; i < NPAD * NPAD; i += 128) {
      const int r = i >> 6, c = i & 63;
      sh_mask[i] = (r < NTOK && c < NTOK) ? maskw[r * NTOK + c] : -1.0e30f;
    }
  }
  __syncthreads();

  const float scale = 0.17677669529663687f;  // 1/sqrt(32)

  // ---- Phase 2: per-head attention ----
  for (int h = 0; h < 4; ++h) {
    const float* biash = biasp + h * (NPAD * NPAD);
    // S^T tiles: A = k rows, B = q (transpose-read). acc(j_local, i_local)
    // = sum_d k[j][d] q[i][d] = S[i][j]; lane covers 8 consecutive j.
    for (int t = wave; t < 16; t += 4) {
      const int mt = t >> 2, nt = t & 3;   // mt: key tile, nt: query tile
      v16bf a    = frag_a (sh_qk + (mt * 16) * 256 + 128 + h * 32, 256, lane);
      v16bf bmat = frag_bt(sh_qk + (nt * 16) * 256 + h * 32, 256, lane);
      v8f acc = {};
      acc = wmma_bf16(a, bmat, acc);
      const int n = lane & 15, kh = lane >> 4;
      const int gi  = nt * 16 + n;            // query index
      const int gjb = mt * 16 + 8 * kh;       // first of 8 consecutive keys
      const float4* bp = (const float4*)(biash + gi * NPAD + gjb);
      const float4* mp = (const float4*)(sh_mask + gi * NPAD + gjb);
      const float4 b0 = bp[0], b1 = bp[1];
      const float4 m0 = mp[0], m1 = mp[1];
      float4 s0, s1;
      s0.x = acc[0] * scale + b0.x + m0.x;
      s0.y = acc[1] * scale + b0.y + m0.y;
      s0.z = acc[2] * scale + b0.z + m0.z;
      s0.w = acc[3] * scale + b0.w + m0.w;
      s1.x = acc[4] * scale + b1.x + m1.x;
      s1.y = acc[5] * scale + b1.y + m1.y;
      s1.z = acc[6] * scale + b1.z + m1.z;
      s1.w = acc[7] * scale + b1.w + m1.w;
      float4* sp = (float4*)(sh_S + gi * NPAD + gjb);
      sp[0] = s0;
      sp[1] = s1;
    }
    __syncthreads();

    // Row softmax: row cached in registers; unnormalized exp stored as bf16.
    if (tid < NPAD) {
      const float4* Sr = (const float4*)(sh_S + tid * NPAD);
      float4 rv[16];
      float mx = -3.4e38f;
#pragma unroll
      for (int j = 0; j < 16; ++j) {
        rv[j] = Sr[j];
        mx = fmaxf(mx, fmaxf(fmaxf(rv[j].x, rv[j].y), fmaxf(rv[j].z, rv[j].w)));
      }
      float s = 0.0f;
      uint2* Pr = (uint2*)(sh_P + tid * NPAD);
#pragma unroll
      for (int j = 0; j < 16; ++j) {
        const float e0 = __expf(rv[j].x - mx), e1 = __expf(rv[j].y - mx);
        const float e2 = __expf(rv[j].z - mx), e3 = __expf(rv[j].w - mx);
        s += (e0 + e1) + (e2 + e3);
        uint2 pk;
        pk.x = (unsigned)f2bf(e0) | ((unsigned)f2bf(e1) << 16);
        pk.y = (unsigned)f2bf(e2) | ((unsigned)f2bf(e3) << 16);
        Pr[j] = pk;
      }
      sh_inv[tid] = 1.0f / s;
    }
    __syncthreads();

    // O_h = (P @ v_h) * inv_rowsum   (64x64 @ 64x32), v^T gives K-contig frags
    for (int t = wave; t < 8; t += 4) {
      const int mt = t >> 1, vt = t & 1;
      v8f acc = {};
#pragma unroll
      for (int kk = 0; kk < 2; ++kk) {
        v16bf a    = frag_a (sh_P + (mt * 16) * NPAD + kk * 32, NPAD, lane);
        v16bf bmat = frag_bt(sh_vT + (h * 32 + vt * 16) * NPAD + kk * 32, NPAD, lane);
        acc = wmma_bf16(a, bmat, acc);
      }
      const int n = lane & 15, kh = lane >> 4;
#pragma unroll
      for (int r = 0; r < 8; ++r) {
        const int row = mt * 16 + r + 8 * kh;
        const float v = acc[r] * sh_inv[row];
        sh_O[row * CDIM + h * 32 + vt * 16 + n] = f2bf(v);
      }
    }
    __syncthreads();
  }

  // ---- Phase 3: out = O @ proj_w + proj_b  (64x128 @ 128x128) ----
  float* outw = out + (size_t)wi * (NTOK * CDIM);
  for (int t = wave; t < 32; t += 4) {
    const int mt = t >> 3, nt = t & 7;
    v8f acc = {};
#pragma unroll
    for (int kk = 0; kk < 4; ++kk) {
      v16bf a    = frag_a(sh_O + (mt * 16) * CDIM + kk * 32, CDIM, lane);
      v16bf bmat = frag_g(projw_bf + (kk * 8 + nt) * 512, lane);
      acc = wmma_bf16(a, bmat, acc);
    }
    const int n = lane & 15, kh = lane >> 4;
    const int gc = nt * 16 + n;
    const float pb = proj_b[gc];
#pragma unroll
    for (int r = 0; r < 8; ++r) {
      const int row = mt * 16 + r + 8 * kh;
      if (row < NTOK) outw[row * CDIM + gc] = acc[r] + pb;
    }
  }
}

extern "C" void kernel_launch(void* const* d_in, const int* in_sizes, int n_in,
                              void* d_out, int out_size, void* d_ws, size_t ws_size,
                              hipStream_t stream) {
  const float* x          = (const float*)d_in[0];
  const float* mask       = (const float*)d_in[1];
  const float* qkv_w      = (const float*)d_in[2];
  const float* qkv_b      = (const float*)d_in[3];
  const float* proj_w     = (const float*)d_in[4];
  const float* proj_b     = (const float*)d_in[5];
  const float* bias_table = (const float*)d_in[6];
  const int*   rel_index  = (const int*)d_in[7];
  float* out = (float*)d_out;
  unsigned short* wbf = (unsigned short*)d_ws;             // 128 KB fragments
  float* biasp = (float*)((char*)d_ws + WS_BIAS_OFF * 2);  // + 64 KB bias

  const int nwin  = in_sizes[0] / (NTOK * CDIM);   // 16384
  const int nmask = in_sizes[1] / NN;              // 1024

  swizzle_weights_kernel<<<16, 256, 0, stream>>>(qkv_w, proj_w, wbf);
  bias_expand_kernel<<<(4 * NPAD * NPAD + 255) / 256, 256, 0, stream>>>(
      bias_table, rel_index, biasp);
  win_attn_kernel<<<nwin, 128, 0, stream>>>(
      x, mask, qkv_b, proj_b, wbf, biasp, out, nmask);
}
